// TemperSeq2Seq_75359496175807
// MI455X (gfx1250) — matrix operand order repaired and checked by reading, not compiled, not run on hardware
//
#include <hip/hip_runtime.h>
#include <hip/hip_bf16.h>

#define B_   32
#define SSRC 256
#define SDEC 128
#define H_   256
#define G3   768
#define VOUT 32000
#define MROWS 4096   // SDEC*B_

typedef __attribute__((ext_vector_type(16))) __bf16         bf16x16;
typedef __attribute__((ext_vector_type(8)))  float          v8f;
typedef __attribute__((ext_vector_type(8)))  unsigned short us8;

#if defined(__AMDGCN__) && __has_builtin(__builtin_amdgcn_global_load_async_to_lds_b128) && __has_builtin(__builtin_amdgcn_s_wait_asynccnt)
#define USE_ASYNC_LDS 1
#endif

#ifdef USE_ASYNC_LDS
// Parameter types per hipcc diagnostic: arg0 = int4* in AS1 (printed "__device__"),
// arg1 = int4* in AS3, then two constant ints (offset, cpol).
typedef __attribute__((vector_size(16))) int v4i_t;
typedef __attribute__((address_space(1))) v4i_t g1_v4i;
typedef __attribute__((address_space(3))) v4i_t l3_v4i;
#endif

// ---------- helpers ----------

__device__ __forceinline__ unsigned short f2bf(float x) {
  union { float f; unsigned int u; } c; c.f = x;
  unsigned int u = c.u;
  unsigned int r = (u + 0x7fffu + ((u >> 16) & 1u)) >> 16;   // RNE
  return (unsigned short)r;
}

// A fragment (16x32 bf16, row-major source, lda in elements).
// Lanes 0-15: row M=m0+lane, K = k0+0..7 and k0+16..23;
// lanes 16-31: row M=m0+lane-16, K = k0+8..15 and k0+24..31.
__device__ __forceinline__ bf16x16 load_frag_a(const unsigned short* A, int lda,
                                               int m0, int k0, int lane) {
  int m  = m0 + (lane & 15);
  int kb = k0 + ((lane & 16) ? 8 : 0);
  const us8* p0 = (const us8*)(A + (size_t)m * lda + kb);
  const us8* p1 = (const us8*)(A + (size_t)m * lda + kb + 16);
  union { us8 s[2]; bf16x16 v; } u;
  u.s[0] = *p0; u.s[1] = *p1;
  return u.v;
}

// B fragment (32x16 bf16) from W stored row-major [n][k] (ldb=K).
// Lanes 0-15: col n0+lane, K=k0+0..7 and k0+16..23; lanes 16-31: col n0+lane-16, K=k0+8..15 and k0+24..31.
__device__ __forceinline__ bf16x16 load_frag_b(const unsigned short* Bm, int ldb,
                                               int n0, int k0, int lane) {
  int n  = n0 + (lane & 15);
  int kb = k0 + ((lane & 16) ? 8 : 0);
  const us8* p0 = (const us8*)(Bm + (size_t)n * ldb + kb);
  const us8* p1 = (const us8*)(Bm + (size_t)n * ldb + kb + 16);
  union { us8 s[2]; bf16x16 v; } u;
  u.s[0] = *p0; u.s[1] = *p1;
  return u.v;
}

__device__ __forceinline__ v8f wmma_bf16(bf16x16 a, bf16x16 b, v8f c) {
  return __builtin_amdgcn_wmma_f32_16x16x32_bf16(false, a, false, b,
                                                 (short)0, c, false, false);
}

// ---------- small kernels ----------

__global__ void pool_kernel(const int* __restrict__ x, const float* __restrict__ emb,
                            float* __restrict__ x_rep) {
  int b = blockIdx.x, h = threadIdx.x;
  const int* xb = x + b * SSRC;
  float s = 0.f;
  for (int i = 0; i < SSRC; i++) s += emb[(size_t)xb[i] * H_ + h];
  x_rep[b * H_ + h] = s * (1.0f / SSRC);
}

__global__ void z_kernel(const float* __restrict__ x_rep, const float* __restrict__ Wt,
                         const float* __restrict__ bt, float* __restrict__ z) {
  __shared__ float xr[H_];
  int b = blockIdx.x, o = threadIdx.x;
  xr[o] = x_rep[b * H_ + o];
  __syncthreads();
  const float* w = Wt + (size_t)o * H_;
  float s = bt[o];
  for (int k = 0; k < H_; k++) s += xr[k] * w[k];
  z[b * H_ + o] = tanhf(s);
}

__global__ void cvt_kernel(const float* __restrict__ in, unsigned short* __restrict__ out, int n) {
  int i = blockIdx.x * 256 + threadIdx.x;
  if (i < n) out[i] = f2bf(in[i]);
}

__global__ void inp_kernel(const int* __restrict__ y_in, const float* __restrict__ dec_emb,
                           const float* __restrict__ z, unsigned short* __restrict__ inp) {
  int row = blockIdx.x;              // row = t*32 + b
  int t = row >> 5, b = row & 31;
  int h = threadIdx.x;
  int tok = y_in[b * SDEC + t];
  inp[(size_t)row * H_ + h] = f2bf(dec_emb[(size_t)tok * H_ + h] + z[b * H_ + h]);
}

// ---------- tiled bf16 WMMA GEMM: D(MxN) = A(MxK) @ Bm(NxK)^T + bias ----------
// WG tile 128(M) x 128(N), K=256. 8 waves, each a 16(M) x 128(N) strip.
// B tile (128 rows x 256 K = 64 KB, contiguous in memory) staged once into LDS
// via CDNA5 async global->LDS DMA (ASYNCcnt / s_wait_asynccnt).
__global__ __launch_bounds__(256) void gemm_bf16(const unsigned short* __restrict__ A,
                                                 const unsigned short* __restrict__ Bm,
                                                 const float* __restrict__ bias,
                                                 float* __restrict__ D,
                                                 int M, int N, int K, int ntn) {
  __shared__ unsigned short Btile[128 * 256];   // 64 KB

  int wg = blockIdx.x;
  int tm = wg / ntn, tn = wg % ntn;
  int wave = threadIdx.x >> 5;
  int lane = threadIdx.x & 31;
  int m0 = tm * 128 + wave * 16;
  int n0 = tn * 128;

  // ---- stage B tile: 4096 chunks of 16B, contiguous 64 KB at Bm + n0*K ----
  const unsigned short* Bsrc = Bm + (size_t)n0 * K;
#ifdef USE_ASYNC_LDS
  for (int c = threadIdx.x; c < 4096; c += 256) {
    g1_v4i* gsrc = (g1_v4i*)(unsigned long long)(Bsrc + (size_t)c * 8);
    l3_v4i* ldst = (l3_v4i*)(unsigned int)(unsigned long long)(Btile + (size_t)c * 8);
    __builtin_amdgcn_global_load_async_to_lds_b128(gsrc, ldst, 0, 0);
  }
  __builtin_amdgcn_s_wait_asynccnt(0);
#else
  for (int c = threadIdx.x; c < 4096; c += 256) {
    us8 v = *(const us8*)(Bsrc + (size_t)c * 8);
    *(us8*)(Btile + (size_t)c * 8) = v;
  }
#endif
  __syncthreads();

  v8f acc[8];
  v8f zero = {0.f, 0.f, 0.f, 0.f, 0.f, 0.f, 0.f, 0.f};
#pragma unroll
  for (int j = 0; j < 8; j++) acc[j] = zero;

  for (int k0 = 0; k0 < K; k0 += 32) {
    bf16x16 a = load_frag_a(A, K, m0, k0, lane);
    if (k0 + 32 < K)   // gfx1250 global_prefetch_b8 for next A strip slice
      __builtin_prefetch(A + (size_t)(m0 + (lane & 15)) * K + k0 + 32, 0, 1);
    // preload all 8 B fragments from LDS, then run the WMMA chain
    bf16x16 bfrag[8];
#pragma unroll
    for (int j = 0; j < 8; j++)
      bfrag[j] = load_frag_b(Btile, 256, j * 16, k0, lane);
#pragma unroll
    for (int j = 0; j < 8; j++)
      acc[j] = wmma_bf16(a, bfrag[j], acc[j]);
  }

  int row_base = m0 + ((lane & 16) ? 8 : 0);
#pragma unroll
  for (int j = 0; j < 8; j++) {
    int n = n0 + j * 16 + (lane & 15);
    float bv = bias[n];
#pragma unroll
    for (int v = 0; v < 8; v++)
      D[(size_t)(row_base + v) * N + n] = acc[j][v] + bv;
  }
}

// ---------- sequential GRU recurrence: one WG, 32 waves, h in LDS/registers ----------
__global__ __launch_bounds__(1024) void gru_recur(const unsigned short* __restrict__ Whh,
                                                  const float* __restrict__ b_hh,
                                                  const float* __restrict__ gx,
                                                  unsigned short* __restrict__ h_all) {
  extern __shared__ char smem[];
  float* gh = (float*)smem;                                            // 32*768 f32 (96 KB)
  unsigned short* h_bf = (unsigned short*)(smem + (size_t)B_ * G3 * 4); // 32*256 bf16 (16 KB)

  int wave = threadIdx.x >> 5;
  int lane = threadIdx.x & 31;

  float hreg[8];
#pragma unroll
  for (int j = 0; j < 8; j++) hreg[j] = 0.f;
  for (int i = threadIdx.x; i < B_ * H_; i += 1024) h_bf[i] = 0;
  __syncthreads();

  for (int t = 0; t < SDEC; t++) {
    // gh = h @ Whh^T + b_hh : M=32, N=768, K=256 -> 96 16x16 tiles, 3 per wave.
    for (int tile = wave; tile < 96; tile += 32) {
      int m0 = (tile & 1) * 16;
      int n0 = (tile >> 1) * 16;
      // preload the whole K chain (8 A frags from LDS + 8 B frags from L2)
      bf16x16 afr[8], bfr[8];
#pragma unroll
      for (int kk = 0; kk < 8; kk++) {
        afr[kk] = load_frag_a(h_bf, H_, m0, kk * 32, lane);
        bfr[kk] = load_frag_b(Whh, H_, n0, kk * 32, lane);
      }
      v8f acc = {0.f, 0.f, 0.f, 0.f, 0.f, 0.f, 0.f, 0.f};
#pragma unroll
      for (int kk = 0; kk < 8; kk++) acc = wmma_bf16(afr[kk], bfr[kk], acc);

      int rb = m0 + ((lane & 16) ? 8 : 0);
      int n  = n0 + (lane & 15);
      float bv = b_hh[n];
#pragma unroll
      for (int v = 0; v < 8; v++) gh[(rb + v) * G3 + n] = acc[v] + bv;
    }
    __syncthreads();

    // elementwise GRU gates (PyTorch order r,z,n); 8 elems per thread
    const float* gxt = gx + (size_t)t * B_ * G3;
#pragma unroll
    for (int j = 0; j < 8; j++) {
      int i = threadIdx.x + j * 1024;
      int b = i >> 8, hh = i & 255;
      const float* gxr = gxt + (size_t)b * G3;
      const float* ghr = gh  + (size_t)b * G3;
      float r  = 1.f / (1.f + __expf(-(gxr[hh]       + ghr[hh])));
      float u  = 1.f / (1.f + __expf(-(gxr[H_ + hh]  + ghr[H_ + hh])));
      float nn = tanhf(gxr[2 * H_ + hh] + r * ghr[2 * H_ + hh]);
      float hn = (1.f - u) * nn + u * hreg[j];
      hreg[j] = hn;
      unsigned short hb = f2bf(hn);
      h_bf[i] = hb;
      h_all[((size_t)b * SDEC + t) * H_ + hh] = hb;  // rows ordered (b,t) for output GEMM
    }
    __syncthreads();
  }
}

// ---------- workspace layout (bytes, 256-aligned) ----------
static const size_t OFF_XREP = 0;          //  32768
static const size_t OFF_Z    = 32768;      //  32768
static const size_t OFF_INP  = 65536;      //  2097152  (4096x256 bf16)
static const size_t OFF_WIH  = 2162688;    //  393216
static const size_t OFF_WHH  = 2555904;    //  393216
static const size_t OFF_WO   = 2949120;    //  16384000
static const size_t OFF_GX   = 19333120;   //  12582912 (4096x768 f32)
static const size_t OFF_HALL = 31916032;   //  2097152  (4096x256 bf16)
// total ~34.0 MB

extern "C" void kernel_launch(void* const* d_in, const int* in_sizes, int n_in,
                              void* d_out, int out_size, void* d_ws, size_t ws_size,
                              hipStream_t stream) {
  const int*   x       = (const int*)d_in[0];
  const int*   y_in    = (const int*)d_in[1];
  const float* emb     = (const float*)d_in[2];
  const float* dec_emb = (const float*)d_in[3];
  const float* Wt      = (const float*)d_in[4];
  const float* bt      = (const float*)d_in[5];
  const float* W_ih    = (const float*)d_in[6];
  const float* W_hh    = (const float*)d_in[7];
  const float* b_ih    = (const float*)d_in[8];
  const float* b_hh    = (const float*)d_in[9];
  const float* Wo      = (const float*)d_in[10];
  const float* bo      = (const float*)d_in[11];
  float* out = (float*)d_out;
  char*  ws  = (char*)d_ws;

  float*          x_rep  = (float*)(ws + OFF_XREP);
  float*          z      = (float*)(ws + OFF_Z);
  unsigned short* inp_bf = (unsigned short*)(ws + OFF_INP);
  unsigned short* wih_bf = (unsigned short*)(ws + OFF_WIH);
  unsigned short* whh_bf = (unsigned short*)(ws + OFF_WHH);
  unsigned short* wo_bf  = (unsigned short*)(ws + OFF_WO);
  float*          gx     = (float*)(ws + OFF_GX);
  unsigned short* h_all  = (unsigned short*)(ws + OFF_HALL);

  // Encoder: mean-pool embeddings, then z = tanh(x_rep @ Wt^T + bt)
  pool_kernel<<<B_, H_, 0, stream>>>(x, emb, x_rep);
  z_kernel<<<B_, H_, 0, stream>>>(x_rep, Wt, bt, z);

  // Weight conversion to bf16 (once; Wo_bf16 = 16.4 MB, L2-resident)
  cvt_kernel<<<(G3 * H_ + 255) / 256, 256, 0, stream>>>(W_ih, wih_bf, G3 * H_);
  cvt_kernel<<<(G3 * H_ + 255) / 256, 256, 0, stream>>>(W_hh, whh_bf, G3 * H_);
  cvt_kernel<<<(VOUT * H_ + 255) / 256, 256, 0, stream>>>(Wo, wo_bf, VOUT * H_);

  // Decoder inputs (teacher forcing): inp[t*32+b] = bf16(dec_emb[y_in[b,t]] + z[b])
  inp_kernel<<<MROWS, H_, 0, stream>>>(y_in, dec_emb, z, inp_bf);

  // gx = inp @ W_ih^T + b_ih : one parallel WMMA GEMM (4096 x 768 x 256)
  gemm_bf16<<<(MROWS / 128) * (G3 / 128), 256, 0, stream>>>(
      inp_bf, wih_bf, b_ih, gx, MROWS, G3, H_, G3 / 128);

  // Sequential recurrence: single 32-wave workgroup, dynamic LDS 112 KB
  size_t smem = (size_t)B_ * G3 * 4 + (size_t)B_ * H_ * 2;
  gru_recur<<<1, 1024, smem, stream>>>(whh_bf, b_hh, gx, h_all);

  // logits = h_all @ Wo^T + bo : 4096 x 32000 x 256 WMMA GEMM -> d_out (B,S,V)
  gemm_bf16<<<(MROWS / 128) * (VOUT / 128), 256, 0, stream>>>(
      h_all, wo_bf, bo, out, MROWS, VOUT, H_, VOUT / 128);
}
